// Actor_60516089200891
// MI455X (gfx1250) — matrix operand order repaired and checked by reading, not compiled, not run on hardware
//
#include <hip/hip_runtime.h>
#include <hip/hip_bf16.h>

typedef __attribute__((ext_vector_type(16))) _Float16 v16h;
typedef __attribute__((ext_vector_type(8)))  float    v8f;
typedef unsigned int v4u  __attribute__((ext_vector_type(4)));
typedef int          v8i_ __attribute__((ext_vector_type(8)));
typedef int          v4i_ __attribute__((ext_vector_type(4)));

#define T_STEPS 32
#define HDIM    64
#define ADIM    8
#define OBSD    512
#define BATCH   4096
#define ROWS    (BATCH * T_STEPS)   // 131072

static __device__ __forceinline__ float frcp(float x) { return __builtin_amdgcn_rcpf(x); }
static __device__ __forceinline__ float fsigmoid(float x) {
    return frcp(1.0f + __expf(-x));
}
static __device__ __forceinline__ float ftanh(float x) {
    x = fminf(fmaxf(x, -15.0f), 15.0f);
    float e = __expf(-2.0f * x);
    return (1.0f - e) * frcp(1.0f + e);
}

static __device__ __forceinline__ v8f wmma_f16(v16h a, v16h b, v8f c) {
    return __builtin_amdgcn_wmma_f32_16x16x32_f16(false, a, false, b, (short)0, c,
                                                  false, false);
}

// ---------------------------------------------------------------------------
// TDM: stage a linear region of `elems` f16 (elems <= 65535) from global into
// LDS at byte offset lds_addr.  D# layout per CDNA5 ISA 8.3/8.4.
// This toolchain exposes the 6-arg builtin (g0, g1, g2, g3, g4-spare, cpol).
// ---------------------------------------------------------------------------
#if __has_builtin(__builtin_amdgcn_tensor_load_to_lds)
#define HAVE_TDM 1
static __device__ __forceinline__ void tdm_load_lds(uint32_t lds_addr,
                                                    const void* gptr,
                                                    uint32_t elems) {
    uint64_t ga = (uint64_t)(uintptr_t)gptr;
    v4u g0;
    g0[0] = 1u;                                   // count=1 (valid descriptor)
    g0[1] = lds_addr;                             // lds_addr (bytes)
    g0[2] = (uint32_t)ga;                         // global_addr[31:0]
    g0[3] = (uint32_t)((ga >> 32) & 0x01FFFFFFu)  // global_addr[56:32]
            | 0x80000000u;                        // type=2 ("image")
    v8i_ g1;
    g1[0] = (int)0x00010000u;                     // data_size=1 -> 2 bytes
    g1[1] = (int)((elems & 0xFFFFu) << 16);       // tensor_dim0[15:0]
    g1[2] = (int)((elems >> 16) | (1u << 16));    // tensor_dim0[31:16], tensor_dim1=1
    g1[3] = (int)((elems & 0xFFFFu) << 16);       // tile_dim0 = elems
    g1[4] = (int)1u;                              // tile_dim1=1, tile_dim2=0
    g1[5] = (int)elems;                           // tensor_dim0_stride[31:0]
    g1[6] = 0;
    g1[7] = 0;
    v4i_ z4 = {0, 0, 0, 0};
    v8i_ z8 = {0, 0, 0, 0, 0, 0, 0, 0};
    __builtin_amdgcn_tensor_load_to_lds(g0, g1, z4, z4, z8, 0);
}
#else
#define HAVE_TDM 0
#endif

// ---------------------------------------------------------------------------
// Weight prep: W0 [512,64] -> W0t f16 N-major [64][512];
//              Wfw/Wbw [128,256] -> Wt f16 N-major [2][256][128]
// ---------------------------------------------------------------------------
__global__ __launch_bounds__(256) void prep_weights(
    const float* __restrict__ W0, const float* __restrict__ Wfw,
    const float* __restrict__ Wbw, _Float16* __restrict__ W0t,
    _Float16* __restrict__ Wt) {
    int idx = blockIdx.x * 256 + threadIdx.x;
    if (idx < HDIM * OBSD) {
        int n = idx / OBSD, k = idx % OBSD;
        W0t[idx] = (_Float16)W0[k * HDIM + n];
    } else {
        int j = idx - HDIM * OBSD;          // 0 .. 65535
        int d = j / (256 * 128);
        int r = j % (256 * 128);
        int n = r / 128, k = r % 128;
        const float* W = d ? Wbw : Wfw;
        Wt[j] = (_Float16)W[k * 256 + n];
    }
}

// ---------------------------------------------------------------------------
// Fused dense(512->64) + layernorm + relu, output f16.
// One wave = 16 rows x 64 cols (4 N-tiles), K-loop of 16 WMMA steps.
// ---------------------------------------------------------------------------
__global__ __launch_bounds__(256) void dense_ln_relu(
    const float* __restrict__ obs, const _Float16* __restrict__ W0t,
    const float* __restrict__ b0, const float* __restrict__ gamma,
    const float* __restrict__ beta, _Float16* __restrict__ x) {
    const int lane  = threadIdx.x & 31;
    const int wave  = blockIdx.x * (blockDim.x >> 5) + (threadIdx.x >> 5);
    const int m0    = wave * 16;
    const int row_l = lane & 15;
    const int kh    = lane >> 4;

    v8f acc[4] = {};
    const float* abase = obs + (size_t)(m0 + row_l) * OBSD + kh * 16;

    for (int ks = 0; ks < 16; ++ks) {
        v16h a;
        const float* p = abase + ks * 32;
#pragma unroll
        for (int i = 0; i < 8; ++i) {
            a[i]     = (_Float16)p[i];
            a[i + 8] = (_Float16)p[8 + i];
        }
#pragma unroll
        for (int nt = 0; nt < 4; ++nt) {
            const _Float16* bp =
                W0t + (size_t)(nt * 16 + row_l) * OBSD + ks * 32 + kh * 16;
            v16h b;
#pragma unroll
            for (int i = 0; i < 16; ++i) b[i] = bp[i];
            acc[nt] = wmma_f16(a, b, acc[nt]);
        }
    }

    float g[4], be[4];
#pragma unroll
    for (int nt = 0; nt < 4; ++nt) {
        float bb = b0[nt * 16 + row_l];
#pragma unroll
        for (int r = 0; r < 8; ++r) acc[nt][r] += bb;
        g[nt]  = gamma[nt * 16 + row_l];
        be[nt] = beta[nt * 16 + row_l];
    }

#pragma unroll
    for (int r = 0; r < 8; ++r) {
        float s = acc[0][r] + acc[1][r] + acc[2][r] + acc[3][r];
        float q = acc[0][r] * acc[0][r] + acc[1][r] * acc[1][r] +
                  acc[2][r] * acc[2][r] + acc[3][r] * acc[3][r];
#pragma unroll
        for (int m = 1; m <= 8; m <<= 1) {
            s += __shfl_xor(s, m, 32);
            q += __shfl_xor(q, m, 32);
        }
        float mu   = s * (1.0f / 64.0f);
        float var  = q * (1.0f / 64.0f) - mu * mu;
        float rstd = rsqrtf(fmaxf(var, 0.0f) + 1e-12f);
        int row = m0 + r + kh * 8;
        _Float16* xrow = x + (size_t)row * HDIM;
#pragma unroll
        for (int nt = 0; nt < 4; ++nt) {
            float v = (acc[nt][r] - mu) * rstd * g[nt] + be[nt];
            xrow[nt * 16 + row_l] = (_Float16)fmaxf(v, 0.0f);
        }
    }
}

// ---------------------------------------------------------------------------
// Persistent bidirectional LSTM.  Block = 256 threads = 8 waves =
// 4 row-groups x 2 col-halves; each row-group owns 16 batch rows for all
// 32 time steps.  Weights staged once into LDS via TDM; h carried in a
// per-row-group LDS double buffer (C-layout store -> A-layout reload);
// c stays in VGPRs.  One workgroup barrier per step; no global sync.
// Grid: 128 blocks (64 per direction).
// ---------------------------------------------------------------------------
__global__ __launch_bounds__(256) void lstm_persistent(
    const _Float16* __restrict__ x, const _Float16* __restrict__ Wt,
    const float* __restrict__ bfw, const float* __restrict__ bbw,
    _Float16* __restrict__ hsbuf) {
    __shared__ _Float16 sW[256 * 128];       // 64 KB: gate weights, N-major
    __shared__ _Float16 sH[4][2][16 * HDIM]; // 16 KB: h double buffer / rowgrp

    const int d    = blockIdx.x >> 6;              // direction
    const int blk  = blockIdx.x & 63;
    const int wave = threadIdx.x >> 5;
    const int lane = threadIdx.x & 31;
    const int rg   = wave >> 1;                    // row group 0..3
    const int ch   = wave & 1;                     // col half 0..1
    const int row_l = lane & 15;
    const int kh    = lane >> 4;
    const int m0    = blk * 64 + rg * 16;          // first batch row

    const _Float16* Wg = Wt + (size_t)d * 256 * 128;
    const float* bias  = d ? bbw : bfw;
    _Float16* hs = hsbuf + (size_t)d * BATCH * T_STEPS * HDIM;

    // --- stage weights into LDS -------------------------------------------
#if HAVE_TDM
    if (threadIdx.x == 0) {
        tdm_load_lds((uint32_t)(uintptr_t)&sW[0], Wg, 256u * 128u);
        __builtin_amdgcn_s_wait_tensorcnt(0);
    }
#else
    for (int i = threadIdx.x; i < 256 * 128 / 8; i += 256)
        ((float4*)sW)[i] = ((const float4*)Wg)[i];
#endif
    // zero h buffer 0 for all row groups
    for (int i = threadIdx.x; i < 4 * 16 * HDIM; i += 256)
        sH[i >> 10][0][i & 1023] = (_Float16)0.0f;
    __syncthreads();

    // per-tile bias (nt = g*2 + cbi -> gate g, col-block ch*2+cbi)
    float bb[8];
#pragma unroll
    for (int nt = 0; nt < 8; ++nt) {
        int g  = nt >> 1;
        int cb = ch * 2 + (nt & 1);
        bb[nt] = bias[g * 64 + cb * 16 + row_l];
    }

    v8f cc[2] = {};     // cell state, 2 tiles of 16x16

    for (int s = 0; s < T_STEPS; ++s) {
        const int t = d ? (T_STEPS - 1 - s) : s;
        const int p = s & 1;

        // A fragments: K 0..63 from x_t, K 64..127 from h (LDS)
        v16h a0, a1, a2, a3;
        {
            const _Float16* xa =
                x + ((size_t)(m0 + row_l) * T_STEPS + t) * HDIM + kh * 16;
            const _Float16* ha = &sH[rg][p][row_l * HDIM + kh * 16];
#pragma unroll
            for (int i = 0; i < 8; ++i) {
                a0[i] = xa[i];       a0[i + 8] = xa[8 + i];
                a1[i] = xa[32 + i];  a1[i + 8] = xa[40 + i];
                a2[i] = ha[i];       a2[i + 8] = ha[8 + i];
                a3[i] = ha[32 + i];  a3[i + 8] = ha[40 + i];
            }
        }

        v8f acc[8] = {};
#pragma unroll
        for (int nt = 0; nt < 8; ++nt) {
            int g  = nt >> 1;
            int cb = ch * 2 + (nt & 1);
            int n0 = g * 64 + cb * 16;
            const _Float16* wb = &sW[(n0 + row_l) * 128 + kh * 16];
            v16h b0, b1, b2, b3;
#pragma unroll
            for (int i = 0; i < 16; ++i) {
                b0[i] = wb[i];
                b1[i] = wb[32 + i];
                b2[i] = wb[64 + i];
                b3[i] = wb[96 + i];
            }
            acc[nt] = wmma_f16(a0, b0, acc[nt]);
            acc[nt] = wmma_f16(a1, b1, acc[nt]);
            acc[nt] = wmma_f16(a2, b2, acc[nt]);
            acc[nt] = wmma_f16(a3, b3, acc[nt]);
        }

        // gates + state update; write h to other LDS buffer + global hs
#pragma unroll
        for (int cbi = 0; cbi < 2; ++cbi) {
            const int col = (ch * 2 + cbi) * 16 + row_l;
#pragma unroll
            for (int r = 0; r < 8; ++r) {
                const int lrow = r + kh * 8;
                float zi = acc[0 + cbi][r] + bb[0 + cbi];
                float zj = acc[2 + cbi][r] + bb[2 + cbi];
                float zf = acc[4 + cbi][r] + bb[4 + cbi];
                float zo = acc[6 + cbi][r] + bb[6 + cbi];
                float c  = cc[cbi][r] * fsigmoid(zf + 1.0f) +
                           fsigmoid(zi) * ftanh(zj);
                cc[cbi][r] = c;
                float h = ftanh(c) * fsigmoid(zo);
                _Float16 hh = (_Float16)h;
                sH[rg][p ^ 1][lrow * HDIM + col] = hh;
                hs[((size_t)(m0 + lrow) * T_STEPS + t) * HDIM + col] = hh;
            }
        }
        __syncthreads();   // h buffer published; old buffer free to overwrite
    }
}

// ---------------------------------------------------------------------------
// Output projection: out = tanh(hs @ Wc + bc), hs rows = 2*B*T, Wc 64x8.
// ---------------------------------------------------------------------------
__global__ __launch_bounds__(256) void proj_tanh(
    const _Float16* __restrict__ hs, const float* __restrict__ Wc,
    const float* __restrict__ bc, float* __restrict__ out) {
    __shared__ float sWc[HDIM * ADIM];
    __shared__ float sb[ADIM];
    int tid = threadIdx.x;
    for (int i = tid; i < HDIM * ADIM; i += 256) sWc[i] = Wc[i];
    if (tid < ADIM) sb[tid] = bc[tid];
    __syncthreads();

    size_t r = (size_t)blockIdx.x * 256 + tid;   // 0 .. 2*B*T-1
    const _Float16* hp = hs + r * HDIM;
    float acc[ADIM];
#pragma unroll
    for (int a = 0; a < ADIM; ++a) acc[a] = sb[a];
    for (int k = 0; k < HDIM; ++k) {
        float hv = (float)hp[k];
#pragma unroll
        for (int a = 0; a < ADIM; ++a) acc[a] = fmaf(hv, sWc[k * ADIM + a], acc[a]);
    }
    float* op = out + r * ADIM;
#pragma unroll
    for (int a = 0; a < ADIM; ++a) op[a] = ftanh(acc[a]);
}

// ---------------------------------------------------------------------------
extern "C" void kernel_launch(void* const* d_in, const int* in_sizes, int n_in,
                              void* d_out, int out_size, void* d_ws,
                              size_t ws_size, hipStream_t stream) {
    const float* obs   = (const float*)d_in[0];
    const float* W0    = (const float*)d_in[1];
    const float* b0    = (const float*)d_in[2];
    const float* gamma = (const float*)d_in[3];
    const float* beta  = (const float*)d_in[4];
    const float* Wfw   = (const float*)d_in[5];
    const float* bfw   = (const float*)d_in[6];
    const float* Wbw   = (const float*)d_in[7];
    const float* bbw   = (const float*)d_in[8];
    const float* Wc    = (const float*)d_in[9];
    const float* bc    = (const float*)d_in[10];

    char* ws = (char*)d_ws;
    size_t off = 0;
    _Float16* x = (_Float16*)(ws + off);
    off += (size_t)ROWS * HDIM * 2;                       // 16.78 MB
    _Float16* hsbuf = (_Float16*)(ws + off);
    off += (size_t)2 * BATCH * T_STEPS * HDIM * 2;        // 33.55 MB
    _Float16* W0t = (_Float16*)(ws + off);
    off += (size_t)HDIM * OBSD * 2;                       // 64 KB
    _Float16* Wt = (_Float16*)(ws + off);
    off += (size_t)2 * 256 * 128 * 2;                     // 128 KB

    prep_weights<<<384, 256, 0, stream>>>(W0, Wfw, Wbw, W0t, Wt);
    dense_ln_relu<<<1024, 256, 0, stream>>>(obs, W0t, b0, gamma, beta, x);
    lstm_persistent<<<128, 256, 0, stream>>>(x, Wt, bfw, bbw, hsbuf);
    proj_tanh<<<1024, 256, 0, stream>>>(hsbuf, Wc, bc, (float*)d_out);
}
